// QRNNLayer_80290118631663
// MI455X (gfx1250) — matrix operand order repaired and checked by reading, not compiled, not run on hardware
//
#include <hip/hip_runtime.h>
#include <hip/hip_bf16.h>

typedef __attribute__((ext_vector_type(16))) __bf16 v16bf;
typedef __attribute__((ext_vector_type(8)))  float  v8f;
typedef __attribute__((ext_vector_type(4)))  int    i4v;

#define B_   8
#define L_   2048
#define C_   1024
#define H_   1024
#define NT   3072     // 3*H
#define KR   3072     // C*K reduction length
#define BM   128
#define BN   256
#define BK   32

#define AS1 __attribute__((address_space(1)))
#define AS3 __attribute__((address_space(3)))
typedef AS1 i4v* gasync_p;   // global src for async DMA
typedef AS3 i4v* lasync_p;   // LDS dst for async DMA

#if __has_builtin(__builtin_amdgcn_global_load_async_to_lds_b128)
#define HAVE_ASYNC_LDS 1
#else
#define HAVE_ASYNC_LDS 0
#endif

__device__ __forceinline__ unsigned short f2bf(float f) {
  unsigned int u = __builtin_bit_cast(unsigned int, f);
  u = (u + 0x7FFFu + ((u >> 16) & 1u)) >> 16;   // round-to-nearest-even
  return (unsigned short)u;
}

__device__ __forceinline__ void wait_async_all() {
#if HAVE_ASYNC_LDS
#if __has_builtin(__builtin_amdgcn_s_wait_asynccnt)
  __builtin_amdgcn_s_wait_asynccnt(0);
  asm volatile("" ::: "memory");
#else
  asm volatile("s_wait_asynccnt 0x0" ::: "memory");
#endif
#endif
}

// ---------------- conversion kernels (one-shot, tiny cost) ----------------
__global__ __launch_bounds__(256) void cvt_x_bf16(const float* __restrict__ x,
                                                  ushort* __restrict__ xb, int n4) {
  int i = blockIdx.x * blockDim.x + threadIdx.x;
  if (i >= n4) return;
  float4 v = ((const float4*)x)[i];
  ushort4 o;
  o.x = f2bf(v.x); o.y = f2bf(v.y); o.z = f2bf(v.z); o.w = f2bf(v.w);
  ((ushort4*)xb)[i] = o;
}

// Bt[n][k*C + c] = conv_w[n][c][k], bf16  (n-major so B-tiles load like A-tiles)
__global__ __launch_bounds__(256) void cvt_w_bf16(const float* __restrict__ w,
                                                  ushort* __restrict__ Bt, int ne) {
  int e = blockIdx.x * blockDim.x + threadIdx.x;
  if (e >= ne) return;
  int n  = e / KR;
  int kc = e - n * KR;
  int k  = kc >> 10;        // / C_
  int c  = kc & (C_ - 1);
  Bt[e] = f2bf(w[(size_t)n * (C_ * 3) + (size_t)c * 3 + k]);
}

// ---------------- causal-conv GEMM with fused bias + activation ----------------
// zfo[m][n], m = b*L+l, n in [0,3H): tanh for n<H, sigmoid otherwise.
// Block tile 128x256, 8 waves, each wave 64x64 (4x4 WMMA frags).
// Global->LDS via async DMA (no VGPR staging, ASYNCcnt-tracked).
__global__ __launch_bounds__(256)
void qrnn_conv_gemm(const ushort* __restrict__ xb,    // [B,L,C] bf16
                    const ushort* __restrict__ Bt,    // [NT,KR] bf16
                    const float*  __restrict__ bias,  // [NT]
                    float* __restrict__ zfo)          // [B*L, NT] f32
{
  __shared__ uint4 As[2][BM * 4];   // 128 rows x 64B (32 bf16 of K)  = 8KB/buf
  __shared__ uint4 Bs[2][BN * 4];   // 256 n-rows x 64B               = 16KB/buf

  const int tid   = threadIdx.x;
  const int wid   = tid >> 5;
  const int lane  = tid & 31;
  const int half  = lane >> 4;      // K-half select per ISA fragment layout
  const int mlan  = lane & 15;
  const int waveM = wid >> 2;       // 2 waves along M -> 64 rows each
  const int waveN = wid & 3;        // 4 waves along N -> 64 cols each

  const int m0 = blockIdx.x * BM;
  const int n0 = blockIdx.y * BN;

  // loader mapping: A = 2 threads/row (32B each), B = 1 thread/row (64B)
  const int arow = tid >> 1;        // 0..127
  const int achk = tid & 1;         // 0/1
  const int brow = tid;             // 0..255

  const int m = m0 + arow;
  const int b = m >> 11;            // / L_
  const int l = m & (L_ - 1);
  const size_t btrow = (size_t)(n0 + brow) * KR;

  v8f acc[4][4];
#pragma unroll
  for (int i = 0; i < 4; ++i)
#pragma unroll
    for (int j = 0; j < 4; ++j) acc[i][j] = v8f{0.f,0.f,0.f,0.f,0.f,0.f,0.f,0.f};

  // issue stage kk into buffer p (async DMA path: no staging registers)
  auto issueStage = [&](int kk, int p) {
    const int k    = kk >> 10;      // tap index 0..2
    const int c0   = kk & (C_ - 1); // channel base (multiple of 32)
    const int lsrc = l + k - 2;     // causal shift
    const int lclamp = (lsrc >= 0) ? lsrc : 0;
    const ushort* ga = xb + ((size_t)b * L_ + lclamp) * C_ + c0 + achk * 16;
    const ushort* gb = Bt + btrow + kk;
    uint4* la = &As[p][arow * 4 + achk * 2];
    uint4* lb = &Bs[p][brow * 4];
#if HAVE_ASYNC_LDS
#pragma unroll
    for (int q = 0; q < 2; ++q)
      __builtin_amdgcn_global_load_async_to_lds_b128(
          (gasync_p)(ga + q * 8), (lasync_p)(la + q), 0, 0);
#pragma unroll
    for (int q = 0; q < 4; ++q)
      __builtin_amdgcn_global_load_async_to_lds_b128(
          (gasync_p)(gb + q * 8), (lasync_p)(lb + q), 0, 0);
#else
    // fallback: register staging with immediate store (short live ranges)
#pragma unroll
    for (int q = 0; q < 2; ++q) la[q] = *(const uint4*)(ga + q * 8);
#pragma unroll
    for (int q = 0; q < 4; ++q) lb[q] = *(const uint4*)(gb + q * 8);
#endif
  };

  // overwrite causally-invalid A rows with zeros (rare: only l < 2)
  auto fixupStage = [&](int kk, int p) {
    const int k = kk >> 10;
    if (l + k - 2 < 0) {
      const uint4 z = make_uint4(0u, 0u, 0u, 0u);
      As[p][arow * 4 + achk * 2 + 0] = z;
      As[p][arow * 4 + achk * 2 + 1] = z;
    }
  };

  union Frag { uint4 u[2]; v16bf v; };

  auto compute = [&](int p) {
    Frag af[4];
#pragma unroll
    for (int i = 0; i < 4; ++i) {
      const int rA = waveM * 64 + i * 16 + mlan;
      af[i].u[0] = As[p][rA * 4 + half];         // K = half*8 .. +7
      af[i].u[1] = As[p][rA * 4 + 2 + half];     // K = 16+half*8 .. +7
    }
#pragma unroll
    for (int j = 0; j < 4; ++j) {
      Frag bf;
      const int rB = waveN * 64 + j * 16 + mlan;
      bf.u[0] = Bs[p][rB * 4 + half];            // K = half*8 .. +7
      bf.u[1] = Bs[p][rB * 4 + 2 + half];        // K = 16+half*8 .. +7
#pragma unroll
      for (int i = 0; i < 4; ++i)
        acc[i][j] = __builtin_amdgcn_wmma_f32_16x16x32_bf16(
            false, af[i].v, false, bf.v, (short)0, acc[i][j], false, false);
    }
  };

  // prologue: stage 0
  issueStage(0, 0);
  wait_async_all();
  fixupStage(0, 0);
  __syncthreads();

  const int NS = KR / BK;           // 96 k-steps
  for (int s = 0; s < NS - 1; ++s) {
    issueStage((s + 1) * BK, (s + 1) & 1);   // DMA next stage (no regs)
    compute(s & 1);                          // 16 ds_load_b128 + 16 WMMA
    wait_async_all();
    fixupStage((s + 1) * BK, (s + 1) & 1);
    __syncthreads();
  }
  compute((NS - 1) & 1);

  // epilogue: bias + activation, store f32
  const int type = n0 >> 10;        // 0:z(tanh) 1:f(sig) 2:o(sig); BN divides H
#pragma unroll
  for (int j = 0; j < 4; ++j) {
    const int n = n0 + waveN * 64 + j * 16 + mlan;
    const float bb = bias[n];
#pragma unroll
    for (int i = 0; i < 4; ++i) {
      const int mbase = m0 + waveM * 64 + i * 16 + half * 8;
#pragma unroll
      for (int r = 0; r < 8; ++r) {
        float v = acc[i][j][r] + bb;
        v = (type == 0) ? tanhf(v) : (1.f / (1.f + __expf(-v)));
        zfo[(size_t)(mbase + r) * NT + n] = v;
      }
    }
  }
}

// ---------------- fo-pool recurrence: one thread per (b,h) ----------------
__global__ __launch_bounds__(256)
void qrnn_fopool(const float* __restrict__ zfo, float* __restrict__ out,
                 float* __restrict__ hlast) {
  const int b = blockIdx.x >> 2;
  const int h = ((blockIdx.x & 3) << 8) + threadIdx.x;
  const float* base = zfo + (size_t)b * L_ * NT + h;
  float* ob = out + (size_t)b * L_ * H_ + h;
  float hv = 0.f;
  for (int l = 0; l < L_; ++l) {
    const float z = base[0];
    const float f = base[H_];
    const float o = base[2 * H_];
    const float c = f * hv + (1.f - f) * z;
    hv = o * c;
    *ob = hv;
    base += NT;
    ob += H_;
  }
  hlast[b * H_ + h] = hv;
}

// ---------------- LayerNorm over H, in place on d_out ----------------
__global__ __launch_bounds__(256)
void qrnn_ln(float* __restrict__ out, const float* __restrict__ gamma,
             const float* __restrict__ beta) {
  __shared__ float r1[256];
  __shared__ float r2[256];
  float* row = out + (size_t)blockIdx.x * H_;
  float v[4];
  float s1 = 0.f, s2 = 0.f;
#pragma unroll
  for (int j = 0; j < 4; ++j) {
    v[j] = row[threadIdx.x + j * 256];
    s1 += v[j];
    s2 += v[j] * v[j];
  }
  r1[threadIdx.x] = s1; r2[threadIdx.x] = s2;
  __syncthreads();
  for (int off = 128; off > 0; off >>= 1) {
    if (threadIdx.x < off) {
      r1[threadIdx.x] += r1[threadIdx.x + off];
      r2[threadIdx.x] += r2[threadIdx.x + off];
    }
    __syncthreads();
  }
  const float mu  = r1[0] * (1.f / H_);
  const float var = r2[0] * (1.f / H_) - mu * mu;
  const float inv = rsqrtf(var + 1e-5f);
#pragma unroll
  for (int j = 0; j < 4; ++j) {
    const int h = threadIdx.x + j * 256;
    row[h] = (v[j] - mu) * inv * gamma[h] + beta[h];
  }
}

extern "C" void kernel_launch(void* const* d_in, const int* in_sizes, int n_in,
                              void* d_out, int out_size, void* d_ws, size_t ws_size,
                              hipStream_t stream) {
  const float* x     = (const float*)d_in[0];
  const float* w     = (const float*)d_in[1];
  const float* bias  = (const float*)d_in[2];
  const float* gamma = (const float*)d_in[3];
  const float* beta  = (const float*)d_in[4];
  float* out = (float*)d_out;

  char* ws = (char*)d_ws;
  const size_t xb_bytes = (size_t)B_ * L_ * C_ * 2;     // 32 MB bf16
  const size_t bt_bytes = (size_t)NT * KR * 2;          // 18 MB bf16
  ushort* xb  = (ushort*)ws;
  ushort* Bt  = (ushort*)(ws + xb_bytes);
  float*  zfo = (float*)(ws + xb_bytes + bt_bytes);     // 192 MB f32
  float*  hlast = out + (size_t)B_ * L_ * H_;

  const int n4 = B_ * L_ * C_ / 4;
  cvt_x_bf16<<<(n4 + 255) / 256, 256, 0, stream>>>(x, xb, n4);
  const int ne = NT * KR;
  cvt_w_bf16<<<(ne + 255) / 256, 256, 0, stream>>>(w, Bt, ne);

  dim3 g((B_ * L_) / BM, NT / BN);   // 128 x 12 tiles
  qrnn_conv_gemm<<<g, 256, 0, stream>>>(xb, Bt, bias, zfo);

  qrnn_fopool<<<B_ * (H_ / 256), 256, 0, stream>>>(zfo, out, hlast);
  qrnn_ln<<<B_ * L_, 256, 0, stream>>>(out, gamma, beta);
}
